// Refine_wtih_pnp_67860483276933
// MI455X (gfx1250) — matrix operand order repaired and checked by reading, not compiled, not run on hardware
//
#include <hip/hip_runtime.h>

// ---------------------------------------------------------------------------
// Refine_with_pnp on MI455X (gfx1250): bf16 WMMA implicit-GEMM convs over a
// zero-padded NHWC feature (branch-free, software-pipelined fragment loads) +
// dual-accumulator f32 WMMA head GEMM + VALU rasterizer.
// ---------------------------------------------------------------------------

typedef __attribute__((ext_vector_type(16))) __bf16 v16bf;
typedef __attribute__((ext_vector_type(8)))  float  v8f;
typedef __attribute__((ext_vector_type(2)))  float  v2f;

#define BS 4
#define CIN 64
#define CMID 256
#define H 256
#define W 256
#define HP 258                    // padded (1px halo)
#define WP 258
#define NPIX (BS * H * W)         // 262144
#define NPOLY 64
#define NUM_POINT 128
#define NPTS 129                  // ct + 128 poly points
#define KDIM (NPTS * CIN)         // 8256
#define NPOLY_OUT 512             // w_poly rows
#define NFUSE 256                 // w_fuse rows (NUM_POINT*2)

// output layout inside d_out (floats)
#define OUT_COARSE 0
#define OUT_MASK   (NPOLY * NUM_POINT * 2)                 // 16384
#define OUT_F      (OUT_MASK + BS * 16 * H * W)            // 16384 + 4194304

// workspace layout (bytes) - all 1KB-aligned
#define FEATP_BYTES (BS * HP * WP * CIN * 2u)               // 34,080,768
#define OFF_FEATP 0u
#define OFF_W1P   34080768u                                 // bf16 [256][576]:      294,912
#define OFF_W2P   34375680u                                 // bf16 [64][256]:        32,768
#define OFF_MAXM  34408448u                                 // f32  [4][256][256]: 1,048,576
#define OFF_F1T   35457024u                                 // bf16 [NPIX][256]: 134,217,728
#define OFF_FP    169674752u                                // f32  [64][8256]:    2,113,536
#define OFF_T1    171788288u                                // f32  [64][512]:       131,072

__device__ __forceinline__ unsigned short f2bf(float x) {
  unsigned int u = __float_as_uint(x);
  unsigned int r = u + 0x7FFFu + ((u >> 16) & 1u);
  return (unsigned short)(r >> 16);
}

union FragBF { v16bf v; uint4 q[2]; };

// --------------------------- layout / pack kernels -------------------------

// zero the padded feature buffer (halo must be 0; ws is poisoned by harness)
__global__ void k_zero_featp(uint4* __restrict__ dst, int n16) {
  int t = blockIdx.x * 256 + threadIdx.x;
  if (t < n16) dst[t] = uint4{0, 0, 0, 0};
}

// NCHW f32 -> padded NHWC bf16 via LDS tile (coalesced both sides)
__global__ __launch_bounds__(256) void k_nchw_to_nhwc(const float* __restrict__ src,
                                                      unsigned short* __restrict__ dst) {
  __shared__ float tile[CIN][65];
  int x0 = blockIdx.x << 6;          // 4 x-tiles of 64
  int y  = blockIdx.y;
  int b  = blockIdx.z;
  int t  = threadIdx.x;
  for (int i = t; i < CIN * 64; i += 256) {
    int ic = i >> 6, x = i & 63;
    tile[ic][x] = src[(((b * CIN + ic) * H + y) << 8) + x0 + x];
  }
  __syncthreads();
  for (int i = t; i < CIN * 64; i += 256) {
    int x = i >> 6, ic = i & 63;
    dst[((size_t)(b * HP + y + 1) * WP + (x0 + x + 1)) * CIN + ic] = f2bf(tile[ic][x]);
  }
}

// w1 (256,64,3,3) OIHW -> bf16 [oc][tap*64+ic]
__global__ void k_pack_w1(const float* __restrict__ w1, unsigned short* __restrict__ w1p) {
  int t = blockIdx.x * 256 + threadIdx.x;
  if (t >= CMID * 576) return;
  int oc = t / 576, k = t - oc * 576;
  int tap = k >> 6, ic = k & 63;
  int ky = tap / 3, kx = tap - ky * 3;
  w1p[t] = f2bf(w1[((oc * CIN + ic) * 3 + ky) * 3 + kx]);
}

// w2 (64,256,1,1) -> bf16 [oc2][ic]
__global__ void k_pack_w2(const float* __restrict__ w2, unsigned short* __restrict__ w2p) {
  int t = blockIdx.x * 256 + threadIdx.x;
  if (t >= CIN * CMID) return;
  w2p[t] = f2bf(w2[t]);
}

// ------------------------------ rasterizer ---------------------------------

// mask_batch[i][ch][y][x]; reproduces reference's literal s = ct_num[i-1] slicing.
__global__ __launch_bounds__(256) void k_pnp(const float* __restrict__ init_polys,
                                             const int* __restrict__ ct_num,
                                             float* __restrict__ mask_out) {
  int y  = blockIdx.x;
  int ch = blockIdx.y;
  int i  = blockIdx.z;
  int x  = threadIdx.x;
  __shared__ float poly[NUM_POINT * 2];
  int s = (i == 0) ? 0 : ct_num[i - 1];
  poly[x] = init_polys[(s + ch) * (NUM_POINT * 2) + x];
  __syncthreads();
  float py = (float)y, px = (float)x;
  bool inside = false;
#pragma unroll 4
  for (int e = 0; e < NUM_POINT; ++e) {
    float ex1 = poly[2 * e], ey1 = poly[2 * e + 1];
    int e2 = (e + 1) & (NUM_POINT - 1);
    float ex2 = poly[2 * e2], ey2 = poly[2 * e2 + 1];
    bool c1 = (ey1 > py) != (ey2 > py);
    float tt = (py - ey1) / (ey2 - ey1 + 1e-9f);
    float xint = ex1 + tt * (ex2 - ex1);
    inside ^= (c1 & (px < xint));
  }
  float v = (ch < ct_num[i]) ? (inside ? 1.0f : 0.0f) : 0.0f;
  mask_out[(((i * 16 + ch) << 8) + y) * 256 + x] = v;
}

__global__ void k_maxmask(const float* __restrict__ mask_out, float* __restrict__ mm) {
  int t = blockIdx.x * 256 + threadIdx.x;      // NPIX threads
  int b = t >> 16, yx = t & 65535;
  float v = 0.0f;
#pragma unroll
  for (int ch = 0; ch < 16; ++ch) v = fmaxf(v, mask_out[((b * 16 + ch) << 16) + yx]);
  mm[t] = v;
}

// --------------------------- conv1: 3x3, 64->256 ---------------------------
// wave tile: 16 oc x 64 pixels; K = 576 in 18 chunks of 32 (tap-major).
// Padded feature => unconditional b128 loads; all fragments of a chunk are
// loaded before the 4 WMMAs so the waits can stagger (partial loadcnt waits).
__global__ __launch_bounds__(256) void k_conv1_wmma(const unsigned short* __restrict__ featP,
                                                    const unsigned short* __restrict__ w1p,
                                                    const float* __restrict__ b1,
                                                    unsigned short* __restrict__ f1T) {
  int wid  = (blockIdx.x << 3) + (threadIdx.x >> 5);
  int lane = threadIdx.x & 31;
  int lh = lane >> 4, ll = lane & 15;
  int octile = wid & 15;
  int pix0   = (wid >> 4) << 6;
  int b = pix0 >> 16, y = (pix0 >> 8) & 255, x0 = pix0 & 255;

  v8f acc[4] = {};
  const unsigned short* arow = w1p + (size_t)((octile << 4) + ll) * 576;

  for (int tap = 0; tap < 9; ++tap) {
    int ky = tap / 3, kx = tap - ky * 3;
    // padded origin: (y+ky, x0+kx) covers y+ky-1 in unpadded coords
    const unsigned short* base =
        featP + ((size_t)(b * HP + y + ky) * WP + (x0 + kx)) * CIN + (lh << 4);
#pragma unroll
    for (int half = 0; half < 2; ++half) {
      int c = tap * 2 + half;
      int abase = (c << 5) + (lh << 4);
      const unsigned short* bh = base + (half << 5);
      // ---- load phase: A + all 4 B fragments in flight ----
      FragBF A, Bf[4];
      A.q[0] = *reinterpret_cast<const uint4*>(arow + abase);
      A.q[1] = *reinterpret_cast<const uint4*>(arow + abase + 8);
      __builtin_prefetch(arow + abase + 32, 0, 1);
#pragma unroll
      for (int sub = 0; sub < 4; ++sub) {
        const unsigned short* bp = bh + (size_t)((sub << 4) + ll) * CIN;
        Bf[sub].q[0] = *reinterpret_cast<const uint4*>(bp);
        Bf[sub].q[1] = *reinterpret_cast<const uint4*>(bp + 8);
      }
      // ---- compute phase: 4 WMMAs, waits stagger per-fragment ----
#pragma unroll
      for (int sub = 0; sub < 4; ++sub) {
        acc[sub] = __builtin_amdgcn_wmma_f32_16x16x32_bf16(false, A.v, false, Bf[sub].v,
                                                           (short)0, acc[sub], false, false);
      }
    }
  }
  // epilogue: +bias, relu, store bf16 NHWC (pairs packed into dword stores)
  int ocb = (octile << 4) + (lh << 3);
#pragma unroll
  for (int sub = 0; sub < 4; ++sub) {
    int pix = pix0 + (sub << 4) + ll;
    unsigned short* orow = f1T + (size_t)pix * CMID + ocb;
#pragma unroll
    for (int j = 0; j < 8; j += 2) {
      float v0 = acc[sub][j]     + b1[ocb + j];
      float v1 = acc[sub][j + 1] + b1[ocb + j + 1];
      v0 = v0 > 0.0f ? v0 : 0.0f;
      v1 = v1 > 0.0f ? v1 : 0.0f;
      unsigned int pk = (unsigned int)f2bf(v0) | ((unsigned int)f2bf(v1) << 16);
      *reinterpret_cast<unsigned int*>(orow + j) = pk;
    }
  }
}

// ------------------- conv2: 1x1, 256->64 + mask-fuse epilogue --------------
__global__ __launch_bounds__(256) void k_conv2_wmma(const unsigned short* __restrict__ f1T,
                                                    const unsigned short* __restrict__ w2p,
                                                    const float* __restrict__ b2,
                                                    const float* __restrict__ maxmask,
                                                    float* __restrict__ fOut) {
  int wid  = (blockIdx.x << 3) + (threadIdx.x >> 5);
  int lane = threadIdx.x & 31;
  int lh = lane >> 4, ll = lane & 15;
  int octile = wid & 3;                 // 4 tiles of 16 -> 64 oc
  int pix0   = (wid >> 2) << 6;
  int b = pix0 >> 16, y = (pix0 >> 8) & 255, x0 = pix0 & 255;

  v8f acc[4] = {};
  const unsigned short* arow = w2p + (size_t)((octile << 4) + ll) * CMID;

  for (int c = 0; c < 8; ++c) {
    int abase = (c << 5) + (lh << 4);
    int koff = (c << 5) + (lh << 4);
    FragBF A, Bf[4];
    A.q[0] = *reinterpret_cast<const uint4*>(arow + abase);
    A.q[1] = *reinterpret_cast<const uint4*>(arow + abase + 8);
#pragma unroll
    for (int sub = 0; sub < 4; ++sub) {
      int pix = pix0 + (sub << 4) + ll;
      const unsigned short* bp = f1T + (size_t)pix * CMID + koff;
      Bf[sub].q[0] = *reinterpret_cast<const uint4*>(bp);
      Bf[sub].q[1] = *reinterpret_cast<const uint4*>(bp + 8);
    }
#pragma unroll
    for (int sub = 0; sub < 4; ++sub) {
      acc[sub] = __builtin_amdgcn_wmma_f32_16x16x32_bf16(false, A.v, false, Bf[sub].v,
                                                         (short)0, acc[sub], false, false);
    }
  }
  int ocb = (octile << 4) + (lh << 3);
#pragma unroll
  for (int sub = 0; sub < 4; ++sub) {
    int n = (sub << 4) + ll;
    int pix = pix0 + n;
    float mm = maxmask[pix];
    float scale = 1.0f + mm;
    int xx = x0 + n;
#pragma unroll
    for (int j = 0; j < 8; ++j) {
      int oc = ocb + j;
      float v = (acc[sub][j] + b2[oc]) * scale;   // relu(mm*f + f)
      v = v > 0.0f ? v : 0.0f;
      fOut[(size_t)(((b * CIN + oc) << 8) + y) * 256 + xx] = v;
    }
  }
}

// --------------------------- bilinear sampling -----------------------------
// fp[p][c*129 + pt] = bilinear(f[idx[p], c], point(p, pt))
__global__ void k_sample(const float* __restrict__ f, const float* __restrict__ ct_polys,
                         const float* __restrict__ init_polys, const int* __restrict__ ct_img_idx,
                         float* __restrict__ fp) {
  int t = blockIdx.x * 256 + threadIdx.x;
  if (t >= NPOLY * NPTS * CIN) return;
  int c = t & 63;
  int rem = t >> 6;
  int pt = rem % NPTS;
  int p  = rem / NPTS;
  int b = ct_img_idx[p];
  float X, Y;
  if (pt == 0) { X = ct_polys[2 * p];  Y = ct_polys[2 * p + 1]; }
  else {
    int q = (p * NUM_POINT + pt - 1) * 2;
    X = init_polys[q]; Y = init_polys[q + 1];
  }
  float ix = X - 0.5f, iy = Y - 0.5f;
  float x0f = floorf(ix), y0f = floorf(iy);
  float wx1 = ix - x0f, wy1 = iy - y0f;
  float wx0 = 1.0f - wx1, wy0 = 1.0f - wy1;
  const float* fc = f + ((size_t)(b * CIN + c) << 16);
  auto g = [&](float yy, float xx) -> float {
    bool valid = (xx >= 0.0f) && (xx <= 255.0f) && (yy >= 0.0f) && (yy <= 255.0f);
    int xc = (int)fminf(fmaxf(xx, 0.0f), 255.0f);
    int yc = (int)fminf(fmaxf(yy, 0.0f), 255.0f);
    float v = fc[(yc << 8) + xc];
    return valid ? v : 0.0f;
  };
  float out = g(y0f, x0f) * (wy0 * wx0) + g(y0f, x0f + 1.0f) * (wy0 * wx1) +
              g(y0f + 1.0f, x0f) * (wy1 * wx0) + g(y0f + 1.0f, x0f + 1.0f) * (wy1 * wx1);
  fp[(size_t)p * KDIM + c * NPTS + pt] = out;
}

// ------------------ head GEMM1: t1[64x512] = fp @ w_poly.T -----------------
// f32 WMMA (full precision), K=8256; two interleaved accumulators break the
// serial WMMA RAW chain (2064 -> 1032 deep), summed in the epilogue.
__global__ __launch_bounds__(256) void k_head_gemm1(const float* __restrict__ fp,
                                                    const float* __restrict__ w_poly,
                                                    float* __restrict__ t1) {
  int wid  = (blockIdx.x << 3) + (threadIdx.x >> 5);   // 128 waves
  int lane = threadIdx.x & 31;
  int lh = lane >> 4, ll = lane & 15;
  int mtile = wid & 3;          // 64 rows -> 4 tiles
  int ntile = wid >> 2;         // 512 cols -> 32 tiles
  const float* arow = fp + (size_t)((mtile << 4) + ll) * KDIM;
  const float* brow = w_poly + (size_t)((ntile << 4) + ll) * KDIM;
  v8f accA = {}, accB = {};
  for (int kc = 0; kc < KDIM / 4; kc += 2) {
    int kb0 = (kc << 2) + (lh << 1);
    v2f a0 = *reinterpret_cast<const v2f*>(arow + kb0);
    v2f b0 = *reinterpret_cast<const v2f*>(brow + kb0);
    v2f a1 = *reinterpret_cast<const v2f*>(arow + kb0 + 4);
    v2f b1 = *reinterpret_cast<const v2f*>(brow + kb0 + 4);
    accA = __builtin_amdgcn_wmma_f32_16x16x4_f32(false, a0, false, b0, (short)0, accA, false, false);
    accB = __builtin_amdgcn_wmma_f32_16x16x4_f32(false, a1, false, b1, (short)0, accB, false, false);
  }
#pragma unroll
  for (int j = 0; j < 8; ++j) {
    int mm = (mtile << 4) + (lh << 3) + j;
    int nn = (ntile << 4) + ll;
    t1[mm * NPOLY_OUT + nn] = accA[j] + accB[j];
  }
}

// ---------- head GEMM2 + finalize: coarse = (t1 @ w_fuse.T + b)*2 + init ---
__global__ void k_head_gemm2(const float* __restrict__ t1, const float* __restrict__ w_fuse,
                             const float* __restrict__ b_fuse, const float* __restrict__ init_polys,
                             float* __restrict__ coarse) {
  int t = blockIdx.x * 256 + threadIdx.x;     // 64*256
  if (t >= NPOLY * NFUSE) return;
  int o = t & 255, p = t >> 8;
  const float* trow = t1 + p * NPOLY_OUT;
  const float* wrow = w_fuse + o * NPOLY_OUT;
  float s = 0.0f;
#pragma unroll 4
  for (int k = 0; k < NPOLY_OUT; ++k) s += trow[k] * wrow[k];
  s += b_fuse[o];
  coarse[t] = s * 2.0f + init_polys[t];       // STRIDE=2, offs layout == init layout
}

// ---------------------------------------------------------------------------

extern "C" void kernel_launch(void* const* d_in, const int* in_sizes, int n_in,
                              void* d_out, int out_size, void* d_ws, size_t ws_size,
                              hipStream_t stream) {
  const float* feature    = (const float*)d_in[0];
  const float* ct_polys   = (const float*)d_in[1];
  const float* init_polys = (const float*)d_in[2];
  const int*   ct_img_idx = (const int*)d_in[3];
  const int*   ct_num     = (const int*)d_in[4];
  const float* w1     = (const float*)d_in[5];
  const float* b1     = (const float*)d_in[6];
  const float* w2     = (const float*)d_in[7];
  const float* b2     = (const float*)d_in[8];
  const float* w_poly = (const float*)d_in[9];
  const float* w_fuse = (const float*)d_in[10];
  const float* b_fuse = (const float*)d_in[11];

  float* out      = (float*)d_out;
  float* coarse   = out + OUT_COARSE;
  float* mask_out = out + OUT_MASK;
  float* fOut     = out + OUT_F;

  char* ws = (char*)d_ws;
  unsigned short* featP = (unsigned short*)(ws + OFF_FEATP);
  unsigned short* w1p   = (unsigned short*)(ws + OFF_W1P);
  unsigned short* w2p   = (unsigned short*)(ws + OFF_W2P);
  float*          maxm  = (float*)(ws + OFF_MAXM);
  unsigned short* f1T   = (unsigned short*)(ws + OFF_F1T);
  float*          fp    = (float*)(ws + OFF_FP);
  float*          t1    = (float*)(ws + OFF_T1);

  // stage 0: zero halo + layout conversions + rasterizer (independent)
  int n16 = (int)(FEATP_BYTES / 16u);
  k_zero_featp<<<(n16 + 255) / 256, 256, 0, stream>>>((uint4*)featP, n16);
  k_nchw_to_nhwc<<<dim3(W / 64, H, BS), 256, 0, stream>>>(feature, featP);
  k_pack_w1<<<(CMID * 576 + 255) / 256, 256, 0, stream>>>(w1, w1p);
  k_pack_w2<<<(CIN * CMID + 255) / 256, 256, 0, stream>>>(w2, w2p);
  k_pnp<<<dim3(H, 16, BS), 256, 0, stream>>>(init_polys, ct_num, mask_out);
  k_maxmask<<<NPIX / 256, 256, 0, stream>>>(mask_out, maxm);

  // stage 1: conv1 (16 oc-tiles x 4096 pixel-tiles = 65536 waves, 8 waves/block)
  k_conv1_wmma<<<65536 / 8, 256, 0, stream>>>(featP, w1p, b1, f1T);

  // stage 2: conv2 + mask-fuse (4 oc-tiles x 4096 pixel-tiles = 16384 waves)
  k_conv2_wmma<<<16384 / 8, 256, 0, stream>>>(f1T, w2p, b2, maxm, fOut);

  // stage 3: sampling + head
  k_sample<<<(NPOLY * NPTS * CIN + 255) / 256, 256, 0, stream>>>(fOut, ct_polys, init_polys,
                                                                 ct_img_idx, fp);
  k_head_gemm1<<<128 / 8, 256, 0, stream>>>(fp, w_poly, t1);
  k_head_gemm2<<<(NPOLY * NFUSE + 255) / 256, 256, 0, stream>>>(t1, w_fuse, b_fuse,
                                                                init_polys, coarse);
}
